// _ManualDecoderLayer_10857677324776
// MI455X (gfx1250) — compile-verified, matrix operand order checked by reading, not run on hardware
//
#include <hip/hip_runtime.h>
#include <hip/hip_bf16.h>

// ---------------- types ----------------
typedef __attribute__((ext_vector_type(16))) __bf16 bf16x16;
typedef __attribute__((ext_vector_type(8)))  float  floatx8;
typedef __attribute__((ext_vector_type(8)))  unsigned short ushort8;
typedef __attribute__((ext_vector_type(4)))  unsigned short ushort4v;

// fp32 -> bf16 (RNE-ish)
__device__ __forceinline__ unsigned short f2bf(float x) {
    unsigned u = __float_as_uint(x);
    unsigned r = u + 0x7FFFu + ((u >> 16) & 1u);
    return (unsigned short)(r >> 16);
}

// LDS byte offset of a generic pointer to __shared__ memory
__device__ __forceinline__ unsigned lds_off(const void* p) {
    return (unsigned)(size_t)(const __attribute__((address_space(3))) void*)p;
}

// CDNA5 async global->LDS copy, 16 bytes per lane, tracked by ASYNCcnt.
__device__ __forceinline__ void async_b128(unsigned loff, const void* g) {
    asm volatile("global_load_async_to_lds_b128 %0, %1, off"
                 :: "v"(loff), "v"(g) : "memory");
}
__device__ __forceinline__ void wait_async_le4() {
    asm volatile("s_wait_asynccnt 0x4" ::: "memory");
}
__device__ __forceinline__ void wait_async_0() {
    asm volatile("s_wait_asynccnt 0x0" ::: "memory");
}

// Load one 16x32 bf16 fragment (A-layout / symmetric B-layout) from LDS.
// Lane L<16 covers row (L), K in {8*half..+8} U {16+8*half..+8}; per ISA
// 16-bit A-matrix striping: two contiguous 8-element runs -> two b128 loads.
__device__ __forceinline__ bf16x16 load_frag(const unsigned short* base,
                                             int row_stride, int kbase) {
    int lane = threadIdx.x & 31;
    int half = lane >> 4;
    int row  = lane & 15;
    const unsigned short* p = base + row * row_stride + kbase + half * 8;
    union { bf16x16 f; ushort8 u[2]; } r;
    r.u[0] = *(const ushort8*)(p);
    r.u[1] = *(const ushort8*)(p + 16);
    return r.f;
}

// B-fragment for P x V from a row-major [K=32 keys][row_stride] LDS tile:
// lane L covers output column (base + L&15), gathering its 16 K elements
// (two runs of 8) as strided 16-bit LDS reads (transpose-in-LDS).
__device__ __forceinline__ bf16x16 load_frag_colT(const unsigned short* base,
                                                  int row_stride) {
    int lane = threadIdx.x & 31;
    int half = lane >> 4;
    int colv = lane & 15;
    union { bf16x16 f; unsigned short u[16]; } r;
#pragma unroll
    for (int j = 0; j < 8; j++) {
        r.u[j]     = base[(8 * half + j) * row_stride + colv];
        r.u[8 + j] = base[(16 + 8 * half + j) * row_stride + colv];
    }
    return r.f;
}

__device__ __forceinline__ floatx8 wmma_bf16(bf16x16 a, bf16x16 b, floatx8 c) {
    return __builtin_amdgcn_wmma_f32_16x16x32_bf16(false, a, false, b,
                                                   (short)0, c, false, false);
}

// ---------------- convert kernel ----------------
__global__ __launch_bounds__(256)
void k_cvt_bf16(const float* __restrict__ src, unsigned short* __restrict__ dst,
                int n4 /* n/4 */) {
    int i = blockIdx.x * blockDim.x + threadIdx.x;
    if (i >= n4) return;
    const float4 v = *(const float4*)(src + i * 4);
    ushort4v o;
    o.x = f2bf(v.x); o.y = f2bf(v.y); o.z = f2bf(v.z); o.w = f2bf(v.w);
    *(ushort4v*)(dst + i * 4) = o;
}

// ---------------- GEMM: C[M,N] = A[M,K] * W[N,K]^T + bias, optional relu ----
// block tile 128(M) x 128(N), K-step 32, 256 threads = 8 waves.
// Each wave: 32x64 sub-tile = 2x4 WMMA accumulators (8 WMMA / K-step).
// Double-buffered LDS fed by async global->LDS DMA (ASYNCcnt pipeline).
__global__ __launch_bounds__(256)
void k_gemm_bf16(const unsigned short* __restrict__ A,
                 const unsigned short* __restrict__ W,
                 const float* __restrict__ bias,
                 float* __restrict__ C,
                 int M, int N, int K, int relu) {
    __shared__ __align__(16) unsigned short As[2][128 * 32];
    __shared__ __align__(16) unsigned short Ws[2][128 * 32];

    const int tid  = threadIdx.x;
    const int m0   = blockIdx.y * 128;
    const int n0   = blockIdx.x * 128;
    const int wave = tid >> 5;
    const int wm   = (wave & 3) * 32;   // 0,32,64,96
    const int wn   = (wave >> 2) * 64;  // 0,64

    // staging: each thread owns 16 elems of each tile (row ar, cols ac..ac+15)
    const int ar = (tid * 16) >> 5;     // 0..127
    const int ac = (tid * 16) & 31;     // 0 or 16
    const unsigned short* Ag = A + (size_t)(m0 + ar) * K + ac;
    const unsigned short* Wg = W + (size_t)(n0 + ar) * K + ac;

    auto issue = [&](int buf, int k0) {
        unsigned la = lds_off(&As[buf][ar * 32 + ac]);
        unsigned lw = lds_off(&Ws[buf][ar * 32 + ac]);
        async_b128(la,      Ag + k0);
        async_b128(la + 16, Ag + k0 + 8);
        async_b128(lw,      Wg + k0);
        async_b128(lw + 16, Wg + k0 + 8);
    };

    floatx8 acc[2][4] = {};
    const int T = K >> 5;

    issue(0, 0);
    for (int t = 0; t < T; t++) {
        if (t + 1 < T) { issue((t + 1) & 1, (t + 1) * 32); wait_async_le4(); }
        else           { wait_async_0(); }
        __syncthreads();

        const unsigned short* Ab = As[t & 1];
        const unsigned short* Wb = Ws[t & 1];
        bf16x16 af[2], wf[4];
        af[0] = load_frag(Ab + (wm + 0)  * 32, 32, 0);
        af[1] = load_frag(Ab + (wm + 16) * 32, 32, 0);
        for (int j = 0; j < 4; j++)
            wf[j] = load_frag(Wb + (wn + 16 * j) * 32, 32, 0);

        for (int i = 0; i < 2; i++)
            for (int j = 0; j < 4; j++)
                acc[i][j] = wmma_bf16(af[i], wf[j], acc[i][j]);
        __syncthreads();
    }

    const int lane = tid & 31;
    const int half = lane >> 4;
    const int col  = lane & 15;
    for (int j = 0; j < 4; j++) {
        int n = n0 + wn + 16 * j + col;
        float bv = bias ? bias[n] : 0.0f;
        for (int i = 0; i < 2; i++) {
            for (int r = 0; r < 8; r++) {
                int m = m0 + wm + 16 * i + r + 8 * half;
                float v = acc[i][j][r] + bv;
                if (relu) v = fmaxf(v, 0.0f);
                C[(size_t)m * N + n] = v;
            }
        }
    }
}

// ---------------- flash attention ----------------
// grid.x = SEQ_Q/64 (q tiles), grid.y = batch*16 + head.
// 128 threads = 4 waves; each wave owns 16 q rows. kv tiles of 32 keys.
// d_k = 64. Strides are element strides: per-seq (s*) and per-batch (b*).
// All of Q/K/V enter LDS via async global->LDS DMA; V is transposed from
// LDS (cheap strided 16-bit reads) when building the PV B-fragment.
__global__ __launch_bounds__(128)
void k_attn(const unsigned short* __restrict__ Qg,
            const unsigned short* __restrict__ Kg,
            const unsigned short* __restrict__ Vg,
            float* __restrict__ Og,
            int sQ, int bQ, int sK, int bK, int sO, int bO,
            int seq_kv, int causal) {
    __shared__ __align__(16) unsigned short Qs[64 * 64];     // q tile
    __shared__ __align__(16) unsigned short Ks[32 * 64];     // key tile [key][d]
    __shared__ __align__(16) unsigned short Vs[32 * 64];     // V tile  [key][d]
    __shared__ __align__(16) unsigned short Ps[4 * 16 * 32]; // per-wave P

    const int q0 = blockIdx.x * 64;
    const int bh = blockIdx.y;
    const int b  = bh >> 4;
    const int h  = bh & 15;
    const unsigned short* Qb = Qg + (size_t)b * bQ + h * 64;
    const unsigned short* Kb = Kg + (size_t)b * bK + h * 64;
    const unsigned short* Vb = Vg + (size_t)b * bK + h * 64;
    float* Ob = Og + (size_t)b * bO + h * 64;

    const int tid  = threadIdx.x;
    const int wave = tid >> 5;
    const int lane = tid & 31;
    const int half = lane >> 4;
    const int col  = lane & 15;

    // stage Q tile (64x64) via async DMA
    for (int e = tid * 8; e < 64 * 64; e += 128 * 8) {
        int r = e >> 6, c = e & 63;
        async_b128(lds_off(&Qs[r * 64 + c]), Qb + (size_t)(q0 + r) * sQ + c);
    }
    wait_async_0();
    __syncthreads();

    float m_run[8], l_run[8];
    floatx8 o[4] = {};
    for (int r = 0; r < 8; r++) { m_run[r] = -1e30f; l_run[r] = 0.0f; }

    const float scale = 0.125f; // 1/sqrt(64)
    int kv_end = causal ? (q0 + 64) : seq_kv;
    if (kv_end > seq_kv) kv_end = seq_kv;

    const int kr = (tid * 16) >> 6;      // K/V-tile staging coords
    const int kc = (tid * 16) & 63;

    for (int t0 = 0; t0 < kv_end; t0 += 32) {
        { // stage K and V tiles (32x64 each) via async DMA
            const unsigned short* gk = Kb + (size_t)(t0 + kr) * sK + kc;
            const unsigned short* gv = Vb + (size_t)(t0 + kr) * sK + kc;
            async_b128(lds_off(&Ks[kr * 64 + kc]),     gk);
            async_b128(lds_off(&Ks[kr * 64 + kc + 8]), gk + 8);
            async_b128(lds_off(&Vs[kr * 64 + kc]),     gv);
            async_b128(lds_off(&Vs[kr * 64 + kc + 8]), gv + 8);
        }
        wait_async_0();
        __syncthreads();

        // S (16 q x 32 keys) per wave: 2 n-tiles, d_k=64 -> 2 k-chunks each
        floatx8 s[2] = {};
        for (int c2 = 0; c2 < 2; c2++) {
            bf16x16 qa  = load_frag(Qs + (wave * 16) * 64, 64, c2 * 32);
            bf16x16 kb0 = load_frag(Ks,           64, c2 * 32);
            bf16x16 kb1 = load_frag(Ks + 16 * 64, 64, c2 * 32);
            s[0] = wmma_bf16(qa, kb0, s[0]);
            s[1] = wmma_bf16(qa, kb1, s[1]);
        }

        // mask + scale + tile row max
        float tmax[8];
        for (int r = 0; r < 8; r++) {
            int qi = q0 + wave * 16 + r + 8 * half;
            float best = -1e30f;
            for (int j = 0; j < 2; j++) {
                int kj = t0 + 16 * j + col;
                float v = s[j][r] * scale;
                if (causal && kj > qi) v = -1e30f;
                s[j][r] = v;
                best = fmaxf(best, v);
            }
            tmax[r] = best;
        }
        for (int msk = 1; msk < 16; msk <<= 1)
            for (int r = 0; r < 8; r++)
                tmax[r] = fmaxf(tmax[r], __shfl_xor(tmax[r], msk, 32));

        // online softmax update
        float corr[8], tsum[8];
        for (int r = 0; r < 8; r++) {
            float mnew = fmaxf(m_run[r], tmax[r]);
            corr[r] = __expf(m_run[r] - mnew);
            m_run[r] = mnew;
            float su = 0.0f;
            for (int j = 0; j < 2; j++) {
                float p = __expf(s[j][r] - mnew);
                s[j][r] = p;
                su += p;
            }
            tsum[r] = su;
        }
        for (int msk = 1; msk < 16; msk <<= 1)
            for (int r = 0; r < 8; r++)
                tsum[r] += __shfl_xor(tsum[r], msk, 32);
        for (int r = 0; r < 8; r++)
            l_run[r] = l_run[r] * corr[r] + tsum[r];
        for (int j = 0; j < 4; j++)
            for (int r = 0; r < 8; r++)
                o[j][r] *= corr[r];

        // P -> per-wave LDS (bf16), re-fragment as A operand
        unsigned short* Pw = &Ps[wave * 16 * 32];
        for (int j = 0; j < 2; j++)
            for (int r = 0; r < 8; r++)
                Pw[(r + 8 * half) * 32 + 16 * j + col] = f2bf(s[j][r]);

        bf16x16 pa = load_frag(Pw, 32, 0);
        for (int n = 0; n < 4; n++) {
            bf16x16 vb = load_frag_colT(Vs + n * 16, 64);
            o[n] = wmma_bf16(pa, vb, o[n]);
        }
        __syncthreads();
    }

    // epilogue: normalize + store fp32
    for (int n = 0; n < 4; n++)
        for (int r = 0; r < 8; r++) {
            int qi = q0 + wave * 16 + r + 8 * half;
            int d  = n * 16 + col;
            Ob[(size_t)qi * sO + d] = o[n][r] / l_run[r];
        }
}

// ---------------- residual + LayerNorm (D=1024, 256 threads/row) ----------
__global__ __launch_bounds__(256)
void k_add_ln(const float* __restrict__ X, const float* __restrict__ Y,
              const float* __restrict__ gw, const float* __restrict__ bw,
              float* __restrict__ O) {
    const int D = 1024;
    const int row = blockIdx.x;
    const int tid = threadIdx.x;
    const float4 xv = *(const float4*)(X + (size_t)row * D + tid * 4);
    const float4 yv = *(const float4*)(Y + (size_t)row * D + tid * 4);
    float t0 = xv.x + yv.x, t1 = xv.y + yv.y, t2 = xv.z + yv.z, t3 = xv.w + yv.w;
    float s  = t0 + t1 + t2 + t3;
    float s2 = t0 * t0 + t1 * t1 + t2 * t2 + t3 * t3;
    for (int msk = 1; msk < 32; msk <<= 1) {
        s  += __shfl_xor(s, msk, 32);
        s2 += __shfl_xor(s2, msk, 32);
    }
    __shared__ float sh[16];
    int wave = tid >> 5, lane = tid & 31;
    if (lane == 0) { sh[wave] = s; sh[8 + wave] = s2; }
    __syncthreads();
    float S = 0.0f, S2 = 0.0f;
    for (int i = 0; i < 8; i++) { S += sh[i]; S2 += sh[8 + i]; }
    float mu  = S / D;
    float inv = rsqrtf(S2 / D - mu * mu + 1e-5f);
    const float4 gv = *(const float4*)(gw + tid * 4);
    const float4 bv = *(const float4*)(bw + tid * 4);
    float4 ov;
    ov.x = (t0 - mu) * inv * gv.x + bv.x;
    ov.y = (t1 - mu) * inv * gv.y + bv.y;
    ov.z = (t2 - mu) * inv * gv.z + bv.z;
    ov.w = (t3 - mu) * inv * gv.w + bv.w;
    *(float4*)(O + (size_t)row * D + tid * 4) = ov;
}

// ---------------- host orchestration ----------------
extern "C" void kernel_launch(void* const* d_in, const int* in_sizes, int n_in,
                              void* d_out, int out_size, void* d_ws, size_t ws_size,
                              hipStream_t stream) {
    (void)in_sizes; (void)n_in; (void)out_size; (void)ws_size;
    const float* tgt  = (const float*)d_in[0];
    const float* mem  = (const float*)d_in[1];
    const float* Wqkv = (const float*)d_in[2];  const float* bqkv = (const float*)d_in[3];
    const float* Wso  = (const float*)d_in[4];  const float* bso  = (const float*)d_in[5];
    const float* Wq   = (const float*)d_in[6];  const float* bq   = (const float*)d_in[7];
    const float* Wk   = (const float*)d_in[8];  const float* bk   = (const float*)d_in[9];
    const float* Wv   = (const float*)d_in[10]; const float* bv   = (const float*)d_in[11];
    const float* Wco  = (const float*)d_in[12]; const float* bco  = (const float*)d_in[13];
    const float* W1   = (const float*)d_in[14]; const float* b1   = (const float*)d_in[15];
    const float* W2   = (const float*)d_in[16]; const float* b2   = (const float*)d_in[17];
    const float* g1   = (const float*)d_in[18]; const float* be1  = (const float*)d_in[19];
    const float* g2   = (const float*)d_in[20]; const float* be2  = (const float*)d_in[21];
    const float* g3   = (const float*)d_in[22]; const float* be3  = (const float*)d_in[23];
    float* out = (float*)d_out;

    const size_t MB = 1u << 20;
    char* ws = (char*)d_ws;
    float* F0 = (float*)(ws + 0);            // 64 MB fp32 scratch (qkv / kc+vc / hidden)
    float* F1 = (float*)(ws + 64 * MB);      // 16 MB fp32 scratch
    float* F2 = (float*)(ws + 80 * MB);      // 16 MB fp32 scratch
    float* X1 = (float*)(ws + 96 * MB);      // LN1 output
    float* X2 = (float*)(ws + 112 * MB);     // LN2 output
    unsigned short* Wb  = (unsigned short*)(ws + 128 * MB); // 8 MB  weight bf16
    unsigned short* Ab  = (unsigned short*)(ws + 136 * MB); // 32 MB activation bf16
    unsigned short* Qb  = (unsigned short*)(ws + 168 * MB); // 8 MB  cross-Q bf16
    unsigned short* KVb = (unsigned short*)(ws + 176 * MB); // 32 MB qkv / k,v bf16

    const int DM = 1024, RQ = 4096, RM = 8192, DFF = 4096;

    auto cvt = [&](const float* s, unsigned short* d, int n) {
        k_cvt_bf16<<<dim3(n / 1024), dim3(256), 0, stream>>>(s, d, n / 4);
    };
    auto gemm = [&](const unsigned short* A, const unsigned short* W,
                    const float* bias, float* C, int M, int N, int K, int relu) {
        k_gemm_bf16<<<dim3(N / 128, M / 128), dim3(256), 0, stream>>>(
            A, W, bias, C, M, N, K, relu);
    };

    // --- self-attention block ---
    cvt(tgt, Ab, RQ * DM);                                  // tgt_bf
    cvt(Wqkv, Wb, 3 * DM * DM);                             // Wqkv_bf
    gemm(Ab, Wb, bqkv, F0, RQ, 3 * DM, DM, 0);              // qkv fp32
    cvt(F0, KVb, RQ * 3 * DM);                              // qkv bf16
    k_attn<<<dim3(16, 64), dim3(128), 0, stream>>>(
        KVb, KVb + 1024, KVb + 2048, F1,
        12288, 3072, 12288, 3072, 4096, 1024, 1024, /*causal=*/1);
    cvt(F1, Ab, RQ * DM);                                   // attn1 bf16
    cvt(Wso, Wb, DM * DM);
    gemm(Ab, Wb, bso, F2, RQ, DM, DM, 0);                   // self out proj
    k_add_ln<<<dim3(RQ), dim3(256), 0, stream>>>(tgt, F2, g1, be1, X1);

    // --- cross-attention block ---
    cvt(X1, Ab, RQ * DM);
    cvt(Wq, Wb, DM * DM);
    gemm(Ab, Wb, bq, F1, RQ, DM, DM, 0);                    // q_c fp32
    cvt(F1, Qb, RQ * DM);                                   // q_c bf16
    cvt(mem, Ab, RM * DM);                                  // memory bf16
    cvt(Wk, Wb, DM * DM);
    gemm(Ab, Wb, bk, F0, RM, DM, DM, 0);                    // k_c fp32
    cvt(Wv, Wb, DM * DM);
    gemm(Ab, Wb, bv, F0 + (size_t)RM * DM, RM, DM, DM, 0);  // v_c fp32
    cvt(F0, KVb, RM * DM);                                  // k_c bf16
    cvt(F0 + (size_t)RM * DM, KVb + (size_t)RM * DM, RM * DM); // v_c bf16
    k_attn<<<dim3(16, 64), dim3(128), 0, stream>>>(
        Qb, KVb, KVb + (size_t)RM * DM, F1,
        4096, 1024, 4096, 1024, 4096, 1024, 2048, /*causal=*/0);
    cvt(F1, Ab, RQ * DM);                                   // attn2 bf16
    cvt(Wco, Wb, DM * DM);
    gemm(Ab, Wb, bco, F2, RQ, DM, DM, 0);                   // cross out proj
    k_add_ln<<<dim3(RQ), dim3(256), 0, stream>>>(X1, F2, g2, be2, X2);

    // --- FFN block ---
    cvt(X2, Ab, RQ * DM);
    cvt(W1, Wb, DFF * DM);
    gemm(Ab, Wb, b1, F0, RQ, DFF, DM, /*relu=*/1);          // hidden fp32
    cvt(F0, Ab, RQ * DFF);                                  // hidden bf16
    cvt(W2, Wb, DM * DFF);
    gemm(Ab, Wb, b2, F1, RQ, DM, DFF, 0);                   // ffn out
    k_add_ln<<<dim3(RQ), dim3(256), 0, stream>>>(X2, F1, g3, be3, out);
}
